// How2comm_74105365725708
// MI455X (gfx1250) — matrix operand order repaired and checked by reading, not compiled, not run on hardware
//
#include <hip/hip_runtime.h>

// ---------------------------------------------------------------------------
// How2comm fusion pipeline for MI455X (gfx1250, wave32, WMMA, async-to-LDS).
// Stages: thetas -> warp(x,conf)->f16 channels-last -> radix-select kth ->
//         masked implicit-GEMM 7x7 conv via v_wmma_f32_16x16x32_f16 -> L-mean.
// ---------------------------------------------------------------------------

#define BB   2
#define LL   5
#define NN   (BB * LL)
#define CCH  64
#define HH_  100
#define WW_  352
#define HW   (HH_ * WW_)
#define KKEEP (HW / 4)

#define WTILE 32
#define PW    (WTILE + 6)   // 38 input columns per tile
#define NWT   11            // ceil(352/32)
#define CIS   72            // padded ci stride in LDS (f16 elems, keeps 16B align)

typedef __attribute__((ext_vector_type(16))) _Float16 v16h;
typedef __attribute__((ext_vector_type(8)))  _Float16 v8h;
typedef __attribute__((ext_vector_type(8)))  float    v8f;

// ---- CDNA5 async global->LDS copy (ASYNCcnt tracked) ----------------------
__device__ __forceinline__ unsigned lds_addr(const void* p) {
    // Shared-aperture generic pointers carry the LDS byte offset in [31:0].
    return (unsigned)(uintptr_t)p;
}
__device__ __forceinline__ void async_g2l_b128(unsigned lds, const void* g) {
    asm volatile("global_load_async_to_lds_b128 %0, %1, off"
                 :: "v"(lds), "v"(g) : "memory");
}
__device__ __forceinline__ void wait_async0() {
    asm volatile("s_wait_asynccnt 0x0" ::: "memory");
}

// ---------------------------------------------------------------------------
// 1) thetas[n][2][3] = pairwise[b,0,l][0:2, (0,1,3)] * scale
// ---------------------------------------------------------------------------
__global__ void theta_kernel(const float* __restrict__ pt, float* __restrict__ thetas) {
    int n = threadIdx.x;
    if (n >= NN) return;
    int b = n / LL, l = n % LL;
    const float s02 = 2.0f / (2.0f * 0.4f * (float)WW_);
    const float s12 = 2.0f / (2.0f * 0.4f * (float)HH_);
    const float* base = pt + ((size_t)(b * LL + 0) * LL + l) * 16;
    thetas[n * 6 + 0] = base[0 * 4 + 0] * 1.0f;
    thetas[n * 6 + 1] = base[0 * 4 + 1] * ((float)HH_ / (float)WW_);
    thetas[n * 6 + 2] = base[0 * 4 + 3] * s02;
    thetas[n * 6 + 3] = base[1 * 4 + 0] * ((float)WW_ / (float)HH_);
    thetas[n * 6 + 4] = base[1 * 4 + 1] * 1.0f;
    thetas[n * 6 + 5] = base[1 * 4 + 3] * s12;
}

// ---------------------------------------------------------------------------
// 2) weights repack: wT[r][ci][co] (f16) from fuse_w[co][ci][r] (f32)
// ---------------------------------------------------------------------------
__global__ void wconv_kernel(const float* __restrict__ fw, _Float16* __restrict__ wT) {
    int i = blockIdx.x * blockDim.x + threadIdx.x;
    if (i >= 49 * 64 * 64) return;
    int co = i & 63;
    int ci = (i >> 6) & 63;
    int r  = i >> 12;
    wT[i] = (_Float16)fw[((size_t)co * 64 + ci) * 49 + r];
}

// ---------------------------------------------------------------------------
// 3) warp: bilinear affine sample of x (64 ch -> f16, channels-last) and of
//    conf = sigmoid(max(psm_ch0, psm_ch1)).
// ---------------------------------------------------------------------------
__global__ void warp_kernel(const float* __restrict__ x, const float* __restrict__ psm,
                            const float* __restrict__ thetas,
                            _Float16* __restrict__ xw, float* __restrict__ confw) {
    int idx = blockIdx.x * blockDim.x + threadIdx.x;
    if (idx >= NN * HW) return;
    int n = idx / HW;
    int p = idx % HW;
    int h = p / WW_;
    int w = p % WW_;

    float t00 = thetas[n * 6 + 0], t01 = thetas[n * 6 + 1], t02 = thetas[n * 6 + 2];
    float t10 = thetas[n * 6 + 3], t11 = thetas[n * 6 + 4], t12 = thetas[n * 6 + 5];

    float gx = ((float)w + 0.5f) * (2.0f / (float)WW_) - 1.0f;
    float gy = ((float)h + 0.5f) * (2.0f / (float)HH_) - 1.0f;
    float sx = t00 * gx + t01 * gy + t02;
    float sy = t10 * gx + t11 * gy + t12;
    float px = (sx + 1.0f) * ((float)WW_ * 0.5f) - 0.5f;
    float py = (sy + 1.0f) * ((float)HH_ * 0.5f) - 0.5f;

    float x0 = floorf(px), y0 = floorf(py);
    float wx1 = px - x0, wx0 = 1.0f - wx1;
    float wy1 = py - y0, wy0 = 1.0f - wy1;

    float ys2[2] = { y0, y0 + 1.0f };
    float xs2[2] = { x0, x0 + 1.0f };
    float wys[2] = { wy0, wy1 };
    float wxs[2] = { wx0, wx1 };

    float cw[4];
    int   off[4];
    int c = 0;
    #pragma unroll
    for (int a = 0; a < 2; a++) {
        #pragma unroll
        for (int q = 0; q < 2; q++) {
            float yy = ys2[a], xx = xs2[q];
            float valid = (yy >= 0.0f && yy <= (float)(HH_ - 1) &&
                           xx >= 0.0f && xx <= (float)(WW_ - 1)) ? 1.0f : 0.0f;
            int yi = (int)fminf(fmaxf(yy, 0.0f), (float)(HH_ - 1));
            int xi = (int)fminf(fmaxf(xx, 0.0f), (float)(WW_ - 1));
            off[c] = yi * WW_ + xi;
            cw[c]  = wys[a] * wxs[q] * valid;
            c++;
        }
    }

    // warped confidence
    const float* psm0 = psm + (size_t)n * 2 * HW;
    float confv = 0.0f;
    #pragma unroll
    for (c = 0; c < 4; c++) {
        float m = fmaxf(psm0[off[c]], psm0[HW + off[c]]);
        float s = 1.0f / (1.0f + expf(-m));
        confv += cw[c] * s;
    }
    confw[(size_t)n * HW + p] = confv;

    // warped features, channels-last f16
    const float* xb = x + (size_t)n * CCH * HW;
    _Float16* xo = xw + (size_t)idx * CCH;
    #pragma unroll 4
    for (int ci = 0; ci < CCH; ci++) {
        const float* xc = xb + (size_t)ci * HW;
        float acc = cw[0] * xc[off[0]] + cw[1] * xc[off[1]] +
                    cw[2] * xc[off[2]] + cw[3] * xc[off[3]];
        xo[ci] = (_Float16)acc;
    }
}

// ---------------------------------------------------------------------------
// 4) exact K-th largest per sample via 4-pass (8-bit) radix select.
// ---------------------------------------------------------------------------
__global__ void topk_kernel(const float* __restrict__ confw, float* __restrict__ kth) {
    __shared__ unsigned hist[256];
    __shared__ unsigned sPrefix, sRemain;
    int n = blockIdx.x;
    int tid = threadIdx.x;
    const float* data = confw + (size_t)n * HW;

    if (tid == 0) { sPrefix = 0u; sRemain = KKEEP; }
    __syncthreads();

    for (int shift = 24; shift >= 0; shift -= 8) {
        hist[tid] = 0u;
        __syncthreads();
        unsigned prefix = sPrefix;
        unsigned hmask = (shift == 24) ? 0u : (0xFFFFFFFFu << (shift + 8));
        for (int i = tid; i < HW; i += 256) {
            unsigned b = __float_as_uint(data[i]);
            unsigned key = (b & 0x80000000u) ? ~b : (b | 0x80000000u);
            if (((key ^ prefix) & hmask) == 0u)
                atomicAdd(&hist[(key >> shift) & 255u], 1u);
        }
        __syncthreads();
        if (tid == 0) {
            unsigned remain = sRemain;
            unsigned cum = 0;
            for (int bin = 255; bin >= 0; bin--) {
                unsigned hb = hist[bin];
                if (cum + hb >= remain) {
                    sPrefix = prefix | ((unsigned)bin << shift);
                    sRemain = remain - cum;
                    break;
                }
                cum += hb;
            }
        }
        __syncthreads();
    }
    if (tid == 0) {
        unsigned key = sPrefix;
        unsigned bits = (key & 0x80000000u) ? (key ^ 0x80000000u) : ~key;
        kth[n] = __uint_as_float(bits);
    }
}

// ---------------------------------------------------------------------------
// 5) masked 7x7 conv as 49 tap-GEMMs (K=64) with v_wmma_f32_16x16x32_f16.
//    Async global->LDS staging (ASYNCcnt) + double-buffered tap weights.
//    Block: 256 threads = 8 waves -> 2(M=32 pixel) x 4(N=64 ch) WMMA tiles.
// ---------------------------------------------------------------------------
__launch_bounds__(256)
__global__ void conv_kernel(const _Float16* __restrict__ xw, const float* __restrict__ confw,
                            const float* __restrict__ kth, const _Float16* __restrict__ wT,
                            const float* __restrict__ bias, float* __restrict__ out) {
    __shared__ alignas(16) _Float16 patch[7 * PW * CIS]; // [dh][wloc][ci(pad 72)]
    __shared__ alignas(16) _Float16 bmat[2][64 * CIS];   // double-buffered [ci][co(pad 72)]

    int blk = blockIdx.x;
    int wt  = blk % NWT;
    int h   = (blk / NWT) % HH_;
    int b   = blk / (NWT * HH_);
    int wbase = wt * WTILE;

    int tid  = threadIdx.x;
    int wave = tid >> 5;
    int lane = tid & 31;
    int mtile = wave & 1;        // 0..1 -> 16-pixel group
    int ntile = wave >> 1;       // 0..3 -> 16-channel group
    int row   = lane & 15;       // A-matrix M row
    int kBase = (lane < 16) ? 0 : 8;

    // per-thread B staging addresses: thread t copies f16 elems [t*16, t*16+16)
    const unsigned bRow = (unsigned)(tid >> 2);
    const unsigned bCol = (unsigned)((tid & 3) * 16);
    const unsigned bDst0 = lds_addr(&bmat[0][0]) + (bRow * CIS + bCol) * 2u;
    const unsigned bDst1 = lds_addr(&bmat[1][0]) + (bRow * CIS + bCol) * 2u;
    const unsigned patchBase = lds_addr(&patch[0]);

    v8f acc = { 0.f, 0.f, 0.f, 0.f, 0.f, 0.f, 0.f, 0.f };

    for (int l = 0; l < LL; l++) {
        int n = b * LL + l;
        bool ego = (l == 0);
        float kv = kth[n];

        // ---- issue async patch loads (clamped addresses; mask applied after) ----
        for (int pos = tid; pos < 7 * PW; pos += 256) {
            int dh = pos / PW;
            int wloc = pos % PW;
            int hin = h + dh - 3;
            int win = wbase + wloc - 3;
            bool inb = (hin >= 0 && hin < HH_ && win >= 0 && win < WW_);
            const _Float16* src = inb
                ? (xw + ((size_t)n * HW + (size_t)hin * WW_ + win) * CCH)
                : xw; // safe in-buffer address; zeroed below
            unsigned dst = patchBase + (unsigned)pos * (CIS * 2u);
            #pragma unroll
            for (int i = 0; i < 8; i++)
                async_g2l_b128(dst + i * 16u, src + i * 8);
        }
        // ---- issue async load of tap-0 weights into buffer 0 ----
        {
            const _Float16* src = wT + (size_t)tid * 16;  // r = 0
            async_g2l_b128(bDst0,       src);
            async_g2l_b128(bDst0 + 16u, src + 8);
        }
        wait_async0();

        // ---- mask / zero pass (same thread owns each position) ----
        for (int pos = tid; pos < 7 * PW; pos += 256) {
            int dh = pos / PW;
            int wloc = pos % PW;
            int hin = h + dh - 3;
            int win = wbase + wloc - 3;
            bool keep = false;
            if (hin >= 0 && hin < HH_ && win >= 0 && win < WW_)
                keep = ego || (confw[(size_t)n * HW + hin * WW_ + win] >= kv);
            if (!keep) {
                uint4 z = { 0u, 0u, 0u, 0u };
                uint4* d = (uint4*)(patch + (size_t)pos * CIS);
                #pragma unroll
                for (int i = 0; i < 8; i++) d[i] = z;
            }
        }
        __syncthreads();

        for (int r = 0; r < 49; r++) {
            int cur = r & 1;
            // ---- prefetch next tap weights into the other buffer (async) ----
            if (r < 48) {
                const _Float16* src = wT + (size_t)(r + 1) * 4096 + (size_t)tid * 16;
                unsigned dst = (cur ? bDst0 : bDst1);
                async_g2l_b128(dst,       src);
                async_g2l_b128(dst + 16u, src + 8);
            }

            int dh = r / 7;
            int dw = r % 7;
            const _Float16* arow = patch + (size_t)(dh * PW + mtile * 16 + row + dw) * CIS;

            #pragma unroll
            for (int kt = 0; kt < 2; kt++) {
                // A fragment: 16-bit A 16x32 layout -> two contiguous 16B reads
                const v8h* ap = (const v8h*)(arow + kt * 32 + kBase);
                v8h alo = ap[0];       // K = kBase .. kBase+7
                v8h ahi = ap[2];       // K = 16+kBase .. 16+kBase+7
                v16h a;
                #pragma unroll
                for (int i = 0; i < 8; i++) { a[i] = alo[i]; a[i + 8] = ahi[i]; }

                // B fragment: N = ntile*16 + lane%16, K = kBase..+7 and 16+kBase..+7 (strided rows)
                const _Float16* bp = &bmat[cur][(size_t)(kt * 32 + kBase) * CIS + ntile * 16 + (lane & 15)];
                v16h bf;
                #pragma unroll
                for (int i = 0; i < 8; i++) { bf[i] = bp[(size_t)i * CIS]; bf[i + 8] = bp[(size_t)(16 + i) * CIS]; }

                acc = __builtin_amdgcn_wmma_f32_16x16x32_f16(
                    false, a, false, bf, (short)0, acc, false, false);
            }
            wait_async0();     // next-tap buffer (this thread's chunk) is resident
            __syncthreads();   // everyone done with bmat[cur] reads + staging visible
        }
    }

    // ---- epilogue: out[b,co,h,w] = acc/L + bias[co] ----
    int co = ntile * 16 + (lane & 15);
    float bv = bias[co];
    int mOff = (lane >= 16) ? 8 : 0;
    #pragma unroll
    for (int e = 0; e < 8; e++) {
        int p = mtile * 16 + e + mOff;
        int w = wbase + p;
        if (w < WW_) {
            out[(((size_t)b * CCH + co) * HH_ + h) * WW_ + w] =
                acc[e] * (1.0f / (float)LL) + bv;
        }
    }
}

// ---------------------------------------------------------------------------
// Launch
// ---------------------------------------------------------------------------
extern "C" void kernel_launch(void* const* d_in, const int* in_sizes, int n_in,
                              void* d_out, int out_size, void* d_ws, size_t ws_size,
                              hipStream_t stream) {
    (void)in_sizes; (void)n_in; (void)out_size; (void)ws_size;

    const float* x   = (const float*)d_in[0];
    const float* psm = (const float*)d_in[1];
    // d_in[2]: record_len (unused, as in reference)
    const float* pt  = (const float*)d_in[3];
    const float* fw  = (const float*)d_in[4];
    const float* fb  = (const float*)d_in[5];
    float* out = (float*)d_out;

    char* ws = (char*)d_ws;
    // ws layout (all offsets 16B aligned):
    //   thetas  f32[10*6]           @ 0
    //   kth     f32[10]             @ 256
    //   confw   f32[10*35200]       @ 512          (1,408,000 B)
    //   wT      f16[49*64*64]       @ 1,408,512    (  401,408 B)
    //   xw      f16[10*35200*64]    @ 1,809,920    (45,056,000 B)
    float*    thetas = (float*)(ws + 0);
    float*    kth    = (float*)(ws + 256);
    float*    confw  = (float*)(ws + 512);
    _Float16* wT     = (_Float16*)(ws + 1408512);
    _Float16* xw     = (_Float16*)(ws + 1809920);

    theta_kernel<<<1, 16, 0, stream>>>(pt, thetas);
    wconv_kernel<<<(49 * 64 * 64 + 255) / 256, 256, 0, stream>>>(fw, wT);
    warp_kernel<<<(NN * HW + 255) / 256, 256, 0, stream>>>(x, psm, thetas, xw, confw);
    topk_kernel<<<NN, 256, 0, stream>>>(confw, kth);
    conv_kernel<<<BB * HH_ * NWT, 256, 0, stream>>>(xw, confw, kth, wT, fb, out);
}
